// GraphStream_31430570672705
// MI455X (gfx1250) — compile-verified, matrix operand order checked by reading, not run on hardware
//
#include <hip/hip_runtime.h>
#include <hip/hip_bf16.h>

// ---------------------------------------------------------------------------
// GraphStream (GCNeXt) forward for MI455X / gfx1250.
// All GEMMs are NT f16 WMMA (A:[M][K], B:[N][K], both f16) with f32 accum.
// Tiles staged to LDS with gfx1250 async-load-to-LDS when available.
// ---------------------------------------------------------------------------

typedef _Float16 half8  __attribute__((ext_vector_type(8)));
typedef _Float16 half16 __attribute__((ext_vector_type(16)));
typedef float    v8f    __attribute__((ext_vector_type(8)));

// exact pointee type expected by the async-to-LDS builtin on this toolchain
typedef int gs_i32x4 __attribute__((vector_size(16)));
typedef __attribute__((address_space(1))) gs_i32x4 gs_gvec_t;  // global int4
typedef __attribute__((address_space(3))) gs_i32x4 gs_lvec_t;  // LDS int4

#if defined(__AMDGCN__) && __has_builtin(__builtin_amdgcn_global_load_async_to_lds_b128)
#define GS_ASYNC 1
#else
#define GS_ASYNC 0
#endif

namespace {
constexpr int Bn = 8, Cc = 256, Tt = 2048, TOPICn = 64;
constexpr int Kk = 6, GCN_Gn = 32, CONV_Gn = 4, Wn = 128;
constexpr float EPS = 1e-5f;

constexpr int GTM = 64;    // block tile M
constexpr int GTN = 128;   // block tile N
constexpr int GTK = 32;    // k-step (16x16x32 f16 WMMA)
}

// 16-byte global->LDS stage: async DMA on gfx1250, sync copy fallback
__device__ __forceinline__ void gs_stage16(const _Float16* g, _Float16* l)
{
#if GS_ASYNC
    __builtin_amdgcn_global_load_async_to_lds_b128(
        (gs_gvec_t*)(unsigned long long)g,
        (gs_lvec_t*)(unsigned)(unsigned long long)l, 0, 0);
#else
    *(half8*)l = *(const half8*)g;
#endif
}

__device__ __forceinline__ void gs_async_wait()
{
#if GS_ASYNC
#if __has_builtin(__builtin_amdgcn_s_wait_asynccnt)
    __builtin_amdgcn_s_wait_asynccnt(0);
#else
    asm volatile("s_wait_asynccnt 0x0" ::: "memory");
#endif
#endif
}

// fragment load: 16 halves per lane (two contiguous 8-half runs per ISA layout)
__device__ __forceinline__ half16 gs_frag(const _Float16* p, int hh)
{
    half16 f;
    ((half8*)&f)[0] = *(const half8*)(p + 8 * hh);
    ((half8*)&f)[1] = *(const half8*)(p + 16 + 8 * hh);
    return f;
}

__device__ __forceinline__ void gs_store_tile(const v8f& acc, float* __restrict__ C,
                                              int ldc, int mbase, int nbase,
                                              int row, int hh,
                                              const float* __restrict__ bias, int relu)
{
    const int col = nbase + row;
    const int r0  = mbase + hh * 8;
#pragma unroll
    for (int r = 0; r < 8; ++r) {
        float v = acc[r] + (bias ? bias[r0 + r] : 0.f);
        if (relu) v = fmaxf(v, 0.f);
        C[(long long)(r0 + r) * ldc + col] = v;
    }
}

// ---------------------------------------------------------------------------
// NT GEMM: C[M,N] = A[M,K] * Bt[N,K]^T (+bias[M]) (+relu), f16 in / f32 out.
// Requires M%64==0, N%128==0, K%32==0, lda/ldb%8==0 (all call sites comply).
// 256 threads = 8 waves as 2(M)x4(N); each wave computes 32x32 via 4 WMMAs.
// ---------------------------------------------------------------------------
__global__ __launch_bounds__(256)
void gs_gemm_nt(const _Float16* __restrict__ A, const _Float16* __restrict__ Bt,
                const float* __restrict__ bias, float* __restrict__ C,
                int M, int N, int Kd, int lda, int ldb, int ldc,
                long long strideB, long long strideC, int relu)
{
    __shared__ __align__(16) _Float16 sA[GTM * GTK];  // [m][k]
    __shared__ __align__(16) _Float16 sB[GTN * GTK];  // [n][k]

    const int z = blockIdx.z;
    Bt += (long long)z * strideB;
    C  += (long long)z * strideC;

    const int m0   = blockIdx.y * GTM;
    const int n0   = blockIdx.x * GTN;
    const int tid  = threadIdx.x;
    const int lane = tid & 31;
    const int wave = tid >> 5;
    const int wm   = (wave >> 2) * 32;   // 0 / 32
    const int wn   = (wave & 3) * 32;    // 0 / 32 / 64 / 96
    const int row  = lane & 15;
    const int hh   = lane >> 4;

    const v8f vz = {0.f, 0.f, 0.f, 0.f, 0.f, 0.f, 0.f, 0.f};
    v8f a00 = vz, a01 = vz, a10 = vz, a11 = vz;

    // per-thread staging assignments (16B segments)
    const int ar = tid >> 2, as = tid & 3;        // A: 64 rows x 4 segs
    const int br0 = tid >> 2, bs0 = tid & 3;      // B: 128 rows x 4 segs, 2/thread
    const int br1 = (tid + 256) >> 2, bs1 = tid & 3;

    for (int k0 = 0; k0 < Kd; k0 += GTK) {
        gs_stage16(&A [(long long)(m0 + ar)  * lda + k0 + as  * 8], &sA[ar  * GTK + as  * 8]);
        gs_stage16(&Bt[(long long)(n0 + br0) * ldb + k0 + bs0 * 8], &sB[br0 * GTK + bs0 * 8]);
        gs_stage16(&Bt[(long long)(n0 + br1) * ldb + k0 + bs1 * 8], &sB[br1 * GTK + bs1 * 8]);
        if (k0 + GTK < Kd)
            __builtin_prefetch(&Bt[(long long)(n0 + br0) * ldb + k0 + GTK], 0, 0);
        gs_async_wait();
        __syncthreads();

        const half16 af0 = gs_frag(&sA[(wm + row)      * GTK], hh);
        const half16 af1 = gs_frag(&sA[(wm + 16 + row) * GTK], hh);
        const half16 bf0 = gs_frag(&sB[(wn + row)      * GTK], hh);
        const half16 bf1 = gs_frag(&sB[(wn + 16 + row) * GTK], hh);

        a00 = __builtin_amdgcn_wmma_f32_16x16x32_f16(false, af0, false, bf0, (short)0, a00, false, false);
        a01 = __builtin_amdgcn_wmma_f32_16x16x32_f16(false, af0, false, bf1, (short)0, a01, false, false);
        a10 = __builtin_amdgcn_wmma_f32_16x16x32_f16(false, af1, false, bf0, (short)0, a10, false, false);
        a11 = __builtin_amdgcn_wmma_f32_16x16x32_f16(false, af1, false, bf1, (short)0, a11, false, false);
        __syncthreads();
    }

    gs_store_tile(a00, C, ldc, m0 + wm,      n0 + wn,      row, hh, bias, relu);
    gs_store_tile(a01, C, ldc, m0 + wm,      n0 + wn + 16, row, hh, bias, relu);
    gs_store_tile(a10, C, ldc, m0 + wm + 16, n0 + wn,      row, hh, bias, relu);
    gs_store_tile(a11, C, ldc, m0 + wm + 16, n0 + wn + 16, row, hh, bias, relu);
}

// ---------------------------------------------------------------------------
// (B,C,T) f32 -> (B,T,C) f16 transpose-convert (GEMM "NT" operand layout)
// ---------------------------------------------------------------------------
__global__ void gs_to_f16_xt(const float* __restrict__ x, _Float16* __restrict__ xt,
                             int B, int C, int T)
{
    long long i = (long long)blockIdx.x * blockDim.x + threadIdx.x;
    long long tot = (long long)B * T * C;
    if (i >= tot) return;
    int c = (int)(i % C);
    int t = (int)((i / C) % T);
    int b = (int)(i / ((long long)T * C));
    xt[i] = (_Float16)x[((long long)b * C + c) * T + t];
}

__global__ void gs_f32_to_f16(const float* __restrict__ s, _Float16* __restrict__ d,
                              long long n)
{
    long long i = (long long)blockIdx.x * blockDim.x + threadIdx.x;
    if (i < n) d[i] = (_Float16)s[i];
}

// ---------------------------------------------------------------------------
// Grouped 3-tap conv1d, padding=1 (f32 in / f32 out)
// ---------------------------------------------------------------------------
__global__ void gs_conv3_grouped(const float* __restrict__ x, const float* __restrict__ w,
                                 const float* __restrict__ bias, float* __restrict__ y,
                                 int Cout, int Cin_g, int groups, int T, int relu)
{
    int t  = blockIdx.x * blockDim.x + threadIdx.x;
    int co = blockIdx.y;
    int b  = blockIdx.z;
    if (t >= T) return;
    int opg = Cout / groups;
    int g   = co / opg;
    int Cin = Cin_g * groups;
    const float* xp = x + ((long long)b * Cin + (long long)g * Cin_g) * T;
    const float* wp = w + (long long)co * Cin_g * 3;
    float acc = bias[co];
    for (int ci = 0; ci < Cin_g; ++ci) {
        const float* xr = xp + (long long)ci * T;
        float w0 = wp[ci * 3 + 0], w1 = wp[ci * 3 + 1], w2 = wp[ci * 3 + 2];
        float xm = (t > 0)     ? xr[t - 1] : 0.f;
        float xc = xr[t];
        float xq = (t < T - 1) ? xr[t + 1] : 0.f;
        acc += w0 * xm + w1 * xc + w2 * xq;
    }
    if (relu) acc = acc > 0.f ? acc : 0.f;
    y[((long long)b * Cout + co) * T + t] = acc;
}

// t2 variant: ReLU fused, output transposed f16 [b][t][co] for the t3 NT GEMM
__global__ void gs_conv3_grouped_tf16(const float* __restrict__ x, const float* __restrict__ w,
                                      const float* __restrict__ bias, _Float16* __restrict__ y,
                                      int Cout, int Cin_g, int groups, int T)
{
    int t  = blockIdx.x * blockDim.x + threadIdx.x;
    int co = blockIdx.y;
    int b  = blockIdx.z;
    if (t >= T) return;
    int opg = Cout / groups;
    int g   = co / opg;
    int Cin = Cin_g * groups;
    const float* xp = x + ((long long)b * Cin + (long long)g * Cin_g) * T;
    const float* wp = w + (long long)co * Cin_g * 3;
    float acc = bias[co];
    for (int ci = 0; ci < Cin_g; ++ci) {
        const float* xr = xp + (long long)ci * T;
        float w0 = wp[ci * 3 + 0], w1 = wp[ci * 3 + 1], w2 = wp[ci * 3 + 2];
        float xm = (t > 0)     ? xr[t - 1] : 0.f;
        float xc = xr[t];
        float xq = (t < T - 1) ? xr[t + 1] : 0.f;
        acc += w0 * xm + w1 * xc + w2 * xq;
    }
    y[((long long)b * T + t) * Cout + co] = (_Float16)fmaxf(acc, 0.f);
}

// ---------------------------------------------------------------------------
// BatchNorm over (B, :, T)
// ---------------------------------------------------------------------------
__global__ __launch_bounds__(256)
void gs_bn_stats(const float* __restrict__ y, float* __restrict__ mean,
                 float* __restrict__ rstd, int B, int C, int T)
{
    __shared__ float ss[256], ss2[256];
    int c = blockIdx.x;
    float s = 0.f, s2 = 0.f;
    int NT = B * T;
    for (int i = threadIdx.x; i < NT; i += 256) {
        int b = i / T, t = i - b * T;
        float v = y[((long long)b * C + c) * T + t];
        s += v; s2 += v * v;
    }
    ss[threadIdx.x] = s; ss2[threadIdx.x] = s2;
    __syncthreads();
    for (int o = 128; o > 0; o >>= 1) {
        if (threadIdx.x < o) { ss[threadIdx.x] += ss[threadIdx.x + o];
                               ss2[threadIdx.x] += ss2[threadIdx.x + o]; }
        __syncthreads();
    }
    if (threadIdx.x == 0) {
        float m = ss[0] / NT;
        float v = ss2[0] / NT - m * m;
        mean[c] = m;
        rstd[c] = rsqrtf(v + EPS);
    }
}

__global__ void gs_bn_apply_relu(const float* __restrict__ y, const float* __restrict__ mean,
                                 const float* __restrict__ rstd, const float* __restrict__ gamma,
                                 const float* __restrict__ beta, float* __restrict__ h,
                                 int C, int T, long long total)
{
    long long i = (long long)blockIdx.x * blockDim.x + threadIdx.x;
    if (i >= total) return;
    int c = (int)((i / T) % C);
    float v = gamma[c] * (y[i] - mean[c]) * rstd[c] + beta[c];
    h[i] = v > 0.f ? v : 0.f;
}

// sq[n] = sum_c x[c,n]^2 (f32 input, per batch)
__global__ void gs_sqnorm(const float* __restrict__ x, float* __restrict__ sq, int C, int T)
{
    int n = blockIdx.x * blockDim.x + threadIdx.x;
    if (n >= T) return;
    float acc = 0.f;
    for (int c = 0; c < C; ++c) {
        float v = x[(long long)c * T + n];
        acc += v * v;
    }
    sq[n] = acc;
}

// ---------------------------------------------------------------------------
// top-K=6 of score(m) = 2*G[n,m] - sq[m]
// ---------------------------------------------------------------------------
__global__ __launch_bounds__(64)
void gs_knn_topk(const float* __restrict__ gram, const float* __restrict__ sq,
                 int* __restrict__ idx, int T)
{
    __shared__ float sv[64 * Kk];
    __shared__ int   si[64 * Kk];
    const int n   = blockIdx.x;
    const int tid = threadIdx.x;
    const float* row = gram + (long long)n * T;

    float bv[Kk]; int bi[Kk];
    for (int j = 0; j < Kk; ++j) { bv[j] = -3.4e38f; bi[j] = 0x7fffffff; }

    for (int m = tid; m < T; m += 64) {
        float s = 2.f * row[m] - sq[m];
        if (s > bv[Kk - 1] || (s == bv[Kk - 1] && m < bi[Kk - 1])) {
            bv[Kk - 1] = s; bi[Kk - 1] = m;
            for (int j = Kk - 1; j > 0; --j) {
                bool sw = (bv[j] > bv[j - 1]) ||
                          (bv[j] == bv[j - 1] && bi[j] < bi[j - 1]);
                if (!sw) break;
                float tv = bv[j]; bv[j] = bv[j - 1]; bv[j - 1] = tv;
                int ti = bi[j]; bi[j] = bi[j - 1]; bi[j - 1] = ti;
            }
        }
    }
    for (int j = 0; j < Kk; ++j) { sv[tid * Kk + j] = bv[j]; si[tid * Kk + j] = bi[j]; }
    __syncthreads();
    if (tid == 0) {
        for (int sel = 0; sel < Kk; ++sel) {
            float best = -3.4e38f; int bidx = 0x7fffffff, bpos = 0;
            for (int i = 0; i < 64 * Kk; ++i) {
                if (sv[i] > best || (sv[i] == best && si[i] < bidx)) {
                    best = sv[i]; bidx = si[i]; bpos = i;
                }
            }
            idx[n * Kk + sel] = bidx;
            sv[bpos] = -3.4e38f; si[bpos] = 0x7fffffff;
        }
    }
}

// ---------------------------------------------------------------------------
// graph feature, written NT: ft[(t*K+k)*Cf + c], f16 (reads f16 xt[t][c])
// ---------------------------------------------------------------------------
__global__ void gs_build_feature_nt(const _Float16* __restrict__ xt, const int* __restrict__ idx,
                                    const float* __restrict__ tf, _Float16* __restrict__ ft,
                                    int C, int T, int K, int Cf)
{
    long long i = (long long)blockIdx.x * blockDim.x + threadIdx.x;
    long long tot = (long long)T * K * Cf;
    if (i >= tot) return;
    int c  = (int)(i % Cf);
    long long nk = i / Cf;
    int t  = (int)(nk / K);
    _Float16 v;
    if (c < C)          v = xt[(long long)idx[nk] * C + c];
    else if (c < 2 * C) v = xt[(long long)t * C + (c - C)];
    else                v = (_Float16)tf[c - 2 * C];
    ft[i] = v;
}

// s2: 32-group 4x4 linear + ReLU; reads S1 f32 [co][n], writes NT f16 [n][co]
__global__ void gs_lin_grouped_tf16(const float* __restrict__ x, const float* __restrict__ w,
                                    const float* __restrict__ bias, _Float16* __restrict__ y,
                                    int N)
{
    int n  = blockIdx.x * blockDim.x + threadIdx.x;
    int co = blockIdx.y;
    if (n >= N) return;
    int g = co >> 2;
    float acc = bias[co];
    const float* wp = w + co * 4;
    const float* xp = x + (long long)(g * 4) * N;
    for (int i = 0; i < 4; ++i) acc += wp[i] * xp[(long long)i * N + n];
    y[(long long)n * Wn + co] = (_Float16)fmaxf(acc, 0.f);
}

// max over neighbor axis k
__global__ void gs_maxk(const float* __restrict__ s3, float* __restrict__ smax,
                        int C, int T, int K)
{
    int i = blockIdx.x * blockDim.x + threadIdx.x;
    if (i >= C * T) return;
    const float* p = s3 + (long long)i * K;
    float m = p[0];
    for (int k = 1; k < K; ++k) m = fmaxf(m, p[k]);
    smax[i] = m;
}

__global__ void gs_combine_relu(const float* __restrict__ t3, const float* __restrict__ xin,
                                const float* __restrict__ smax, float* __restrict__ xout,
                                long long total)
{
    long long i = (long long)blockIdx.x * blockDim.x + threadIdx.x;
    if (i >= total) return;
    float v = t3[i] + xin[i] + smax[i];
    xout[i] = v > 0.f ? v : 0.f;
}

__global__ void gs_add_ip(float* __restrict__ a, const float* __restrict__ b, long long total)
{
    long long i = (long long)blockIdx.x * blockDim.x + threadIdx.x;
    if (i >= total) return;
    a[i] += b[i];
}

// (B,T,C) -> (B,C,T)
__global__ void gs_transpose_in(const float* __restrict__ vf, float* __restrict__ x,
                                int B, int T, int C)
{
    long long i = (long long)blockIdx.x * blockDim.x + threadIdx.x;
    long long tot = (long long)B * C * T;
    if (i >= tot) return;
    int b = (int)(i / ((long long)C * T));
    long long rem = i - (long long)b * C * T;
    int c = (int)(rem / T), t = (int)(rem % T);
    x[i] = vf[((long long)b * T + t) * C + c];
}

// out(B,T,C) = transpose(base + ident)
__global__ void gs_final_out(const float* __restrict__ base, const float* __restrict__ ident,
                             float* __restrict__ out, int B, int T, int C)
{
    long long i = (long long)blockIdx.x * blockDim.x + threadIdx.x;
    long long tot = (long long)B * T * C;
    if (i >= tot) return;
    int b = (int)(i / ((long long)T * C));
    long long rem = i - (long long)b * T * C;
    int t = (int)(rem / C), c = (int)(rem % C);
    long long src = ((long long)b * C + c) * T + t;
    out[i] = base[src] + ident[src];
}

// topic embedding: grouped 1x1 conv + BN over batch + relu
__global__ __launch_bounds__(256)
void gs_topic(const float* __restrict__ te, const float* __restrict__ w,
              const float* __restrict__ bias, const float* __restrict__ gamma,
              const float* __restrict__ beta, float* __restrict__ tf)
{
    int c = blockIdx.x * blockDim.x + threadIdx.x;
    if (c >= Cc) return;
    int g = c / (Cc / CONV_Gn);
    const int ipg = TOPICn / CONV_Gn;   // 16
    float v[Bn]; float mean = 0.f;
    for (int b = 0; b < Bn; ++b) {
        float acc = bias[c];
        for (int i = 0; i < ipg; ++i)
            acc += w[c * ipg + i] * te[b * TOPICn + g * ipg + i];
        v[b] = acc; mean += acc;
    }
    mean /= (float)Bn;
    float var = 0.f;
    for (int b = 0; b < Bn; ++b) { float d = v[b] - mean; var += d * d; }
    var /= (float)Bn;
    float rstd = rsqrtf(var + EPS);
    for (int b = 0; b < Bn; ++b) {
        float o = gamma[c] * (v[b] - mean) * rstd + beta[c];
        tf[b * Cc + c] = o > 0.f ? o : 0.f;
    }
}

// ---------------------------------------------------------------------------
// Host orchestration
// ---------------------------------------------------------------------------
namespace {
inline int cdiv(int a, int b) { return (a + b - 1) / b; }

struct GcnW {
    const float *t1b, *t2w, *t2b, *t3b, *s1b, *s2w, *s2b, *s3b;
    _Float16 *t1w16, *t3w16, *s1w16, *s3w16;
    int cin;
};

struct Bufs {
    float *T1, *T3, *GRAM, *SQ, *S1, *S3, *SMX;
    _Float16 *XT, *T2T, *FT, *S2T;
    int* IDX;
};

void run_gcnext(const float* xin, float* xout, const GcnW& g, const float* tf,
                const Bufs& bf, hipStream_t st)
{
    const int NK = Tt * Kk;                 // 12288
    long long total = (long long)Bn * Cc * Tt;
    int eblk = (int)((total + 255) / 256);

    // f16 NT copy of the input: XT[b][t][c]
    gs_to_f16_xt<<<eblk, 256, 0, st>>>(xin, bf.XT, Bn, Cc, Tt);

    // t-chain
    gs_gemm_nt<<<dim3(Tt / GTN, Wn / GTM, Bn), 256, 0, st>>>(
        g.t1w16, bf.XT, g.t1b, bf.T1, Wn, Tt, Cc, Cc, Cc, Tt,
        (long long)Tt * Cc, (long long)Wn * Tt, 1);
    gs_conv3_grouped_tf16<<<dim3(cdiv(Tt, 256), Wn, Bn), 256, 0, st>>>(
        bf.T1, g.t2w, g.t2b, bf.T2T, Wn, Wn / GCN_Gn, GCN_Gn, Tt);
    gs_gemm_nt<<<dim3(Tt / GTN, Cc / GTM, Bn), 256, 0, st>>>(
        g.t3w16, bf.T2T, g.t3b, bf.T3, Cc, Tt, Wn, Wn, Wn, Tt,
        (long long)Tt * Wn, (long long)Cc * Tt, 0);

    // graph path, per batch (Gram stays L2-resident)
    for (int b = 0; b < Bn; ++b) {
        const float*    xb  = xin   + (long long)b * Cc * Tt;
        const _Float16* xtb = bf.XT + (long long)b * Tt * Cc;
        gs_sqnorm<<<cdiv(Tt, 256), 256, 0, st>>>(xb, bf.SQ, Cc, Tt);
        gs_gemm_nt<<<dim3(Tt / GTN, Tt / GTM, 1), 256, 0, st>>>(
            xtb, xtb, nullptr, bf.GRAM, Tt, Tt, Cc, Cc, Cc, Tt, 0LL, 0LL, 0);
        gs_knn_topk<<<Tt, 64, 0, st>>>(bf.GRAM, bf.SQ, bf.IDX, Tt);
        long long ftot = (long long)NK * g.cin;
        gs_build_feature_nt<<<(int)((ftot + 255) / 256), 256, 0, st>>>(
            xtb, bf.IDX, tf ? tf + (long long)b * Cc : nullptr, bf.FT, Cc, Tt, Kk, g.cin);
        gs_gemm_nt<<<dim3(NK / GTN, Wn / GTM, 1), 256, 0, st>>>(
            g.s1w16, bf.FT, g.s1b, bf.S1, Wn, NK, g.cin, g.cin, g.cin, NK, 0LL, 0LL, 1);
        gs_lin_grouped_tf16<<<dim3(cdiv(NK, 256), Wn), 256, 0, st>>>(
            bf.S1, g.s2w, g.s2b, bf.S2T, NK);
        gs_gemm_nt<<<dim3(NK / GTN, Cc / GTM, 1), 256, 0, st>>>(
            g.s3w16, bf.S2T, g.s3b, bf.S3, Cc, NK, Wn, Wn, Wn, NK, 0LL, 0LL, 0);
        gs_maxk<<<cdiv(Cc * Tt, 256), 256, 0, st>>>(
            bf.S3, bf.SMX + (long long)b * Cc * Tt, Cc, Tt, Kk);
    }
    gs_combine_relu<<<eblk, 256, 0, st>>>(bf.T3, xin, bf.SMX, xout, total);
}

void conv_bn_relu(const float* xin, const float* bw, const float* bb,
                  const float* bng, const float* bnb,
                  float* Y, float* MEAN, float* RSTD, float* hout, hipStream_t st)
{
    gs_conv3_grouped<<<dim3(cdiv(Tt, 256), Cc, Bn), 256, 0, st>>>(
        xin, bw, bb, Y, Cc, Cc / CONV_Gn, CONV_Gn, Tt, 0);
    gs_bn_stats<<<Cc, 256, 0, st>>>(Y, MEAN, RSTD, Bn, Cc, Tt);
    long long total = (long long)Bn * Cc * Tt;
    gs_bn_apply_relu<<<(int)((total + 255) / 256), 256, 0, st>>>(
        Y, MEAN, RSTD, bng, bnb, hout, Cc, Tt, total);
}
} // namespace

extern "C" void kernel_launch(void* const* d_in, const int* in_sizes, int n_in,
                              void* d_out, int out_size, void* d_ws, size_t ws_size,
                              hipStream_t stream)
{
    (void)in_sizes; (void)n_in; (void)out_size; (void)ws_size;

    const float* vf = (const float*)d_in[0];   // (B,T,C)
    const float* te = (const float*)d_in[1];   // (B,TOPIC)
    const float* snip_bw  = (const float*)d_in[2];
    const float* snip_bb  = (const float*)d_in[3];
    const float* snip_bng = (const float*)d_in[4];
    const float* snip_bnb = (const float*)d_in[5];
    const float* ego_bw   = (const float*)d_in[30];
    const float* ego_bb   = (const float*)d_in[31];
    const float* ego_bng  = (const float*)d_in[32];
    const float* ego_bnb  = (const float*)d_in[33];
    const float* ego_tw   = (const float*)d_in[34];
    const float* ego_tb   = (const float*)d_in[35];
    const float* ego_tng  = (const float*)d_in[36];
    const float* ego_tnb  = (const float*)d_in[37];

    float* out = (float*)d_out;

    // ---- workspace carving ----
    float* w = (float*)d_ws;
    size_t off = 0;
    auto alloc = [&](size_t n) { float* p = w + off; off += n; return p; };
    const size_t BCT = (size_t)Bn * Cc * Tt;
    float* X0 = alloc(BCT);
    float* P0 = alloc(BCT);
    float* P1 = alloc(BCT);
    float* P2 = alloc(BCT);
    float* Y  = alloc(BCT);
    Bufs bf;
    bf.T3   = alloc(BCT);
    bf.SMX  = alloc(BCT);
    bf.T1   = alloc((size_t)Bn * Wn * Tt);
    bf.GRAM = alloc((size_t)Tt * Tt);
    bf.S1   = alloc((size_t)Wn * Tt * Kk);
    bf.S3   = alloc((size_t)Cc * Tt * Kk);
    bf.SQ   = alloc(Tt);
    float* MEAN = alloc(Cc);
    float* RSTD = alloc(Cc);
    float* TF   = alloc((size_t)Bn * Cc);

    _Float16* hws = (_Float16*)(w + off);
    size_t hoff = 0;
    auto halloc = [&](size_t n) { _Float16* p = hws + hoff; hoff += n; return p; };
    bf.XT  = halloc((size_t)Bn * Tt * Cc);
    bf.T2T = halloc((size_t)Bn * Tt * Wn);
    bf.FT  = halloc((size_t)3 * Cc * Tt * Kk);
    bf.S2T = halloc((size_t)Tt * Kk * Wn);

    // per-gcnext f16 weight buffers + descriptor
    auto cvt = [&](const float* s, _Float16* d, size_t n) {
        gs_f32_to_f16<<<(int)((n + 255) / 256), 256, 0, stream>>>(s, d, (long long)n);
    };
    auto load_gcn = [&](int base, int cin) {
        GcnW g;
        const float* t1w = (const float*)d_in[base + 0];
        g.t1b = (const float*)d_in[base + 1];
        g.t2w = (const float*)d_in[base + 2];
        g.t2b = (const float*)d_in[base + 3];
        const float* t3w = (const float*)d_in[base + 4];
        g.t3b = (const float*)d_in[base + 5];
        const float* s1w = (const float*)d_in[base + 6];
        g.s1b = (const float*)d_in[base + 7];
        g.s2w = (const float*)d_in[base + 8];
        g.s2b = (const float*)d_in[base + 9];
        const float* s3w = (const float*)d_in[base + 10];
        g.s3b = (const float*)d_in[base + 11];
        g.cin = cin;
        g.t1w16 = halloc((size_t)Wn * Cc);
        g.t3w16 = halloc((size_t)Cc * Wn);
        g.s1w16 = halloc((size_t)Wn * cin);
        g.s3w16 = halloc((size_t)Cc * Wn);
        cvt(t1w, g.t1w16, (size_t)Wn * Cc);
        cvt(t3w, g.t3w16, (size_t)Cc * Wn);
        cvt(s1w, g.s1w16, (size_t)Wn * cin);
        cvt(s3w, g.s3w16, (size_t)Cc * Wn);
        return g;
    };
    GcnW snip_g1 = load_gcn(6,  2 * Cc);
    GcnW snip_g2 = load_gcn(18, 2 * Cc);
    GcnW ego_g1  = load_gcn(38, 3 * Cc);
    GcnW ego_g2  = load_gcn(50, 3 * Cc);

    if (hoff & 1) hoff++;                   // 4B-align the int region
    bf.IDX = (int*)(hws + hoff);

    long long total = (long long)BCT;
    int eblk = (int)((total + 255) / 256);

    // --- snip branch ---
    gs_transpose_in<<<eblk, 256, 0, stream>>>(vf, X0, Bn, Tt, Cc);
    conv_bn_relu(X0, snip_bw, snip_bb, snip_bng, snip_bnb, Y, MEAN, RSTD, P0, stream);
    run_gcnext(P0, P1, snip_g1, nullptr, bf, stream);
    run_gcnext(P1, P0, snip_g2, nullptr, bf, stream);
    gs_add_ip<<<eblk, 256, 0, stream>>>(P0, X0, total);   // h = h + x (P0 = identity)

    // --- ego branch ---
    conv_bn_relu(P0, ego_bw, ego_bb, ego_bng, ego_bnb, Y, MEAN, RSTD, P1, stream);
    gs_topic<<<1, 256, 0, stream>>>(te, ego_tw, ego_tb, ego_tng, ego_tnb, TF);
    run_gcnext(P1, P2, ego_g1, TF, bf, stream);
    run_gcnext(P2, P1, ego_g2, TF, bf, stream);

    // --- out = swapaxes(base + identity) ---
    gs_final_out<<<eblk, 256, 0, stream>>>(P1, P0, out, Bn, Tt, Cc);
}
